// GraphNeuralODE_11622181503405
// MI455X (gfx1250) — compile-verified, matrix operand order checked by reading, not run on hardware
//
#include <hip/hip_runtime.h>
#include <hip/hip_bf16.h>
#include <math.h>

typedef __attribute__((ext_vector_type(16))) _Float16 v16h;
typedef __attribute__((ext_vector_type(8)))  float    v8f;
typedef __attribute__((ext_vector_type(4)))  _Float16 v4h;

static constexpr int B_   = 4;
static constexpr int N_   = 25000;
static constexpr int T_   = 96;
static constexpr int BN   = B_ * N_;                 // 100000 nodes
static constexpr int BNP  = ((BN + 63) / 64) * 64;   // 100032, padded for 64-row gemm tiles
static constexpr int HOR  = 10;
static constexpr int F    = 64;

// ---------------------------------------------------------------------------
// WMMA fragment loader: both A (16x32 f16) and B (32x16 f16) fragments have the
// same per-lane K pattern (ISA 7.12.2): lanes 0-15 hold K = kb+0..7 and
// kb+16..23; lanes 16-31 hold K = kb+8..15 and kb+24..31, for row M=lane%16
// (A) / col N=lane%16 (B, loaded from K-major i.e. transposed W).
// ---------------------------------------------------------------------------
__device__ __forceinline__ v16h load_frag_kmajor(const _Float16* __restrict__ vec,
                                                 int kb, int lane) {
  const int off = kb + ((lane & 16) ? 8 : 0);
  union { v16h v; uint4 u[2]; } f;
  f.u[0] = *(const uint4*)(vec + off);        // 8 halves: K = off..off+7
  f.u[1] = *(const uint4*)(vec + off + 16);   // 8 halves: K = off+16..off+23
  return f.v;
}

// Z(BNPx64,f16) = H(BNPx64,f16) @ W(64x64) with W pre-transposed to Wt[n][k].
// 128 threads = 4 waves; each wave computes a 16-row x 64-col strip with
// 8x v_wmma_f32_16x16x32_f16 (4 col tiles x 2 K-steps).
__global__ void __launch_bounds__(128)
k_gemm64_wmma(const _Float16* __restrict__ H, const _Float16* __restrict__ Wt,
              _Float16* __restrict__ Z) {
  const int lane    = threadIdx.x & 31;
  const int wave    = threadIdx.x >> 5;
  const int rowBase = blockIdx.x * 64 + wave * 16;

  const _Float16* hrow = H + (size_t)(rowBase + (lane & 15)) * F;
  const v16h a0 = load_frag_kmajor(hrow, 0,  lane);
  const v16h a1 = load_frag_kmajor(hrow, 32, lane);

  const int rOff = rowBase + ((lane & 16) ? 8 : 0);   // C/D: VGPR j -> row j (+8 hi lanes)
#pragma unroll
  for (int cs = 0; cs < 4; ++cs) {
    const _Float16* wcol = Wt + (size_t)(cs * 16 + (lane & 15)) * F;
    const v16h b0 = load_frag_kmajor(wcol, 0,  lane);
    const v16h b1 = load_frag_kmajor(wcol, 32, lane);
    v8f c = {};
    c = __builtin_amdgcn_wmma_f32_16x16x32_f16(false, a0, false, b0, (short)0, c, false, false);
    c = __builtin_amdgcn_wmma_f32_16x16x32_f16(false, a1, false, b1, (short)0, c, false, false);
    const int n = cs * 16 + (lane & 15);
#pragma unroll
    for (int j = 0; j < 8; ++j)
      Z[(size_t)(rOff + j) * F + n] = (_Float16)c[j];
  }
}

// --------------------------- setup kernels ---------------------------------
__global__ void k_fill_f32(float* __restrict__ p, long long n, float v) {
  long long i = (long long)blockIdx.x * blockDim.x + threadIdx.x;
  if (i < n) p[i] = v;
}

__global__ void k_deg(const int* __restrict__ col, unsigned* __restrict__ deg, int E) {
  int e = blockIdx.x * blockDim.x + threadIdx.x;
  if (e < E) atomicAdd(&deg[col[e]], 1u);
}

__global__ void k_dinv(const unsigned* __restrict__ deg, float* __restrict__ dinv) {
  int i = blockIdx.x * blockDim.x + threadIdx.x;
  if (i < BN) dinv[i] = rsqrtf((float)(deg[i] + 1u));   // +1 self loop
}

__global__ void k_norm(const int* __restrict__ row, const int* __restrict__ col,
                       const float* __restrict__ dinv, float* __restrict__ norm, int E) {
  int e = blockIdx.x * blockDim.x + threadIdx.x;
  if (e < E) norm[e] = dinv[row[e]] * dinv[col[e]];
}

__global__ void k_wt_conv(const float* __restrict__ W, _Float16* __restrict__ Wt) {
  int gid = blockIdx.x * blockDim.x + threadIdx.x;    // 4096
  if (gid < F * F) {
    int k = gid >> 6, n = gid & 63;                   // coalesced read of W[k][n]
    Wt[n * F + k] = (_Float16)W[gid];
  }
}

__global__ void k_init_y(const float* __restrict__ x, float* __restrict__ y) {
  int i = blockIdx.x * blockDim.x + threadIdx.x;
  if (i < BN) y[i] = x[(size_t)i * T_ + (T_ - 1)];
}

__global__ void k_pad_zero(_Float16* __restrict__ Hh) {
  int t = blockIdx.x * blockDim.x + threadIdx.x;
  if (t < (BNP - BN) * F) Hh[(size_t)BN * F + t] = (_Float16)0.f;
}

// --------------------------- GCN layer kernels -----------------------------
// Layer 1: Z[i][j] = y[i] * W1[j]   (1 -> 64 outer product)
__global__ void k_z_outer(const float* __restrict__ y, const float* __restrict__ W1,
                          _Float16* __restrict__ Z) {
  int gid = blockIdx.x * blockDim.x + threadIdx.x;
  if (gid < BN * F) Z[gid] = (_Float16)(y[gid >> 6] * W1[gid & 63]);
}

// Edge aggregation, 64-wide features: 16 threads/edge, 4 features each.
// Gather from L2-resident f16 Z (one 8B load), scatter f32 atomics into acc.
__global__ void k_agg64(const int* __restrict__ row, const int* __restrict__ col,
                        const float* __restrict__ norm, const _Float16* __restrict__ Z,
                        float* __restrict__ acc, int E) {
  long long gid = (long long)blockIdx.x * blockDim.x + threadIdx.x;
  if (gid >= (long long)E * 16) return;
  const int e  = (int)(gid >> 4);
  const int fb = ((int)gid & 15) << 2;
  const int r = row[e], c = col[e];
  const float nm = norm[e];
  const v4h z = *(const v4h*)(Z + (size_t)r * F + fb);
  float* ap = acc + (size_t)c * F + fb;
  atomicAdd(ap + 0, nm * (float)z[0]);
  atomicAdd(ap + 1, nm * (float)z[1]);
  atomicAdd(ap + 2, nm * (float)z[2]);
  atomicAdd(ap + 3, nm * (float)z[3]);
}

// Finalize: self-loop term + bias + tanh, write next-layer H (f16).
__global__ void k_fin64(const float* __restrict__ acc, const _Float16* __restrict__ Z,
                        const float* __restrict__ dinv, const float* __restrict__ b,
                        _Float16* __restrict__ H) {
  int gid = blockIdx.x * blockDim.x + threadIdx.x;
  if (gid >= BN * F) return;
  const int i = gid >> 6, j = gid & 63;
  const float di = dinv[i];
  const float v = acc[gid] + di * di * (float)Z[gid] + b[j];
  H[gid] = (_Float16)tanhf(v);
}

// Layer 5: Zs[i] = dot(H[i,:], W5)   (64 -> 1)
__global__ void k_matvec64(const _Float16* __restrict__ H, const float* __restrict__ W5,
                           float* __restrict__ Zs) {
  int i = blockIdx.x * blockDim.x + threadIdx.x;
  if (i >= BN) return;
  const v4h* hp = (const v4h*)(H + (size_t)i * F);
  float s = 0.f;
#pragma unroll
  for (int t = 0; t < 16; ++t) {
    v4h h4 = hp[t];
    s += (float)h4[0] * W5[4 * t + 0] + (float)h4[1] * W5[4 * t + 1]
       + (float)h4[2] * W5[4 * t + 2] + (float)h4[3] * W5[4 * t + 3];
  }
  Zs[i] = s;
}

__global__ void k_agg1(const int* __restrict__ row, const int* __restrict__ col,
                       const float* __restrict__ norm, const float* __restrict__ Zs,
                       float* __restrict__ acc1, int E) {
  int e = blockIdx.x * blockDim.x + threadIdx.x;
  if (e < E) atomicAdd(&acc1[col[e]], norm[e] * Zs[row[e]]);
}

__global__ void k_fin1(const float* __restrict__ acc1, const float* __restrict__ Zs,
                       const float* __restrict__ dinv, const float* __restrict__ b5,
                       float* __restrict__ kout) {
  int i = blockIdx.x * blockDim.x + threadIdx.x;
  if (i >= BN) return;
  const float di = dinv[i];
  kout[i] = acc1[i] + di * di * Zs[i] + b5[0];
}

// ----------------------------- RK4 glue ------------------------------------
__global__ void k_axpy(const float* __restrict__ y, const float* __restrict__ k,
                       float a, float* __restrict__ out) {
  int i = blockIdx.x * blockDim.x + threadIdx.x;
  if (i < BN) out[i] = y[i] + a * k[i];
}

__global__ void k_combine(float* __restrict__ y, const float* __restrict__ k1,
                          const float* __restrict__ k2, const float* __restrict__ k3,
                          const float* __restrict__ k4, float c) {
  int i = blockIdx.x * blockDim.x + threadIdx.x;
  if (i < BN) y[i] = y[i] + c * (k1[i] + 2.f * k2[i] + 2.f * k3[i] + k4[i]);
}

__global__ void k_store_pred(const float* __restrict__ y, float* __restrict__ out, int h) {
  int i = blockIdx.x * blockDim.x + threadIdx.x;
  if (i < BN) out[(size_t)i * HOR + h] = y[i];
}

// ----------------------------- host side -----------------------------------
static inline int gblk(long long n, int b) { return (int)((n + b - 1) / b); }

namespace {
struct Ctx {
  const int *row, *col;
  const float *W1, *b1, *b2, *b3, *b4, *W5, *b5;
  const _Float16 *Wt2, *Wt3, *Wt4;
  _Float16 *Hh, *Zh;
  float *acc, *norm, *dinv, *Zs, *acc1;
  int E;
};

void ode_f(const Ctx& c, const float* ycur, float* kout, hipStream_t s) {
  const int TB = 256;
  const long long NF = (long long)BN * F;
  // layer 1 (1 -> 64)
  k_z_outer<<<gblk(NF, TB), TB, 0, s>>>(ycur, c.W1, c.Zh);
  k_fill_f32<<<gblk(NF, TB), TB, 0, s>>>(c.acc, NF, 0.f);
  k_agg64<<<gblk((long long)c.E * 16, TB), TB, 0, s>>>(c.row, c.col, c.norm, c.Zh, c.acc, c.E);
  k_fin64<<<gblk(NF, TB), TB, 0, s>>>(c.acc, c.Zh, c.dinv, c.b1, c.Hh);
  // layers 2..4 (64 -> 64), WMMA gemm
  const _Float16* Wts[3] = {c.Wt2, c.Wt3, c.Wt4};
  const float*    bs[3]  = {c.b2, c.b3, c.b4};
  for (int l = 0; l < 3; ++l) {
    k_gemm64_wmma<<<BNP / 64, 128, 0, s>>>(c.Hh, Wts[l], c.Zh);
    k_fill_f32<<<gblk(NF, TB), TB, 0, s>>>(c.acc, NF, 0.f);
    k_agg64<<<gblk((long long)c.E * 16, TB), TB, 0, s>>>(c.row, c.col, c.norm, c.Zh, c.acc, c.E);
    k_fin64<<<gblk(NF, TB), TB, 0, s>>>(c.acc, c.Zh, c.dinv, bs[l], c.Hh);
  }
  // layer 5 (64 -> 1)
  k_matvec64<<<gblk(BN, TB), TB, 0, s>>>(c.Hh, c.W5, c.Zs);
  k_fill_f32<<<gblk(BN, TB), TB, 0, s>>>(c.acc1, BN, 0.f);
  k_agg1<<<gblk(c.E, TB), TB, 0, s>>>(c.row, c.col, c.norm, c.Zs, c.acc1, c.E);
  k_fin1<<<gblk(BN, TB), TB, 0, s>>>(c.acc1, c.Zs, c.dinv, c.b5, kout);
}
} // namespace

extern "C" void kernel_launch(void* const* d_in, const int* in_sizes, int n_in,
                              void* d_out, int out_size, void* d_ws, size_t ws_size,
                              hipStream_t stream) {
  const float* x  = (const float*)d_in[0];
  const int*   ei = (const int*)d_in[1];
  const float* W1 = (const float*)d_in[2];
  const float* b1 = (const float*)d_in[3];
  const float* W2 = (const float*)d_in[4];
  const float* b2 = (const float*)d_in[5];
  const float* W3 = (const float*)d_in[6];
  const float* b3 = (const float*)d_in[7];
  const float* W4 = (const float*)d_in[8];
  const float* b4 = (const float*)d_in[9];
  const float* W5 = (const float*)d_in[10];
  const float* b5 = (const float*)d_in[11];
  float* out = (float*)d_out;

  const int E = in_sizes[1] / 2;
  const int* row = ei;       // edge_index[0] = source
  const int* col = ei + E;   // edge_index[1] = target

  // workspace bump allocator (256B aligned), ~62 MB total
  char* wsp = (char*)d_ws;
  auto alloc = [&](size_t bytes) {
    char* p = wsp; wsp += (bytes + 255) & ~(size_t)255; return (void*)p;
  };
  _Float16* Hh   = (_Float16*)alloc((size_t)BNP * F * 2);
  _Float16* Zh   = (_Float16*)alloc((size_t)BNP * F * 2);
  float*    acc  = (float*)   alloc((size_t)BNP * F * 4);
  float*    norm = (float*)   alloc((size_t)E * 4);
  float*    dinv = (float*)   alloc((size_t)BN * 4);
  unsigned* deg  = (unsigned*)alloc((size_t)BN * 4);
  float*    y    = (float*)   alloc((size_t)BN * 4);
  float*    ytmp = (float*)   alloc((size_t)BN * 4);
  float*    k1   = (float*)   alloc((size_t)BN * 4);
  float*    k2   = (float*)   alloc((size_t)BN * 4);
  float*    k3   = (float*)   alloc((size_t)BN * 4);
  float*    k4   = (float*)   alloc((size_t)BN * 4);
  float*    Zs   = (float*)   alloc((size_t)BN * 4);
  float*    acc1 = (float*)   alloc((size_t)BN * 4);
  _Float16* Wt2  = (_Float16*)alloc((size_t)F * F * 2);
  _Float16* Wt3  = (_Float16*)alloc((size_t)F * F * 2);
  _Float16* Wt4  = (_Float16*)alloc((size_t)F * F * 2);

  const int TB = 256;
  // --- one-time setup: normalization, weight transpose/convert, y0, padding ---
  k_fill_f32<<<gblk(BN, TB), TB, 0, stream>>>((float*)deg, BN, 0.f);  // 0 bits
  k_deg<<<gblk(E, TB), TB, 0, stream>>>(col, deg, E);
  k_dinv<<<gblk(BN, TB), TB, 0, stream>>>(deg, dinv);
  k_norm<<<gblk(E, TB), TB, 0, stream>>>(row, col, dinv, norm, E);
  k_wt_conv<<<gblk(F * F, TB), TB, 0, stream>>>(W2, Wt2);
  k_wt_conv<<<gblk(F * F, TB), TB, 0, stream>>>(W3, Wt3);
  k_wt_conv<<<gblk(F * F, TB), TB, 0, stream>>>(W4, Wt4);
  k_init_y<<<gblk(BN, TB), TB, 0, stream>>>(x, y);
  k_pad_zero<<<gblk((BNP - BN) * F, TB), TB, 0, stream>>>(Hh);

  Ctx c{row, col, W1, b1, b2, b3, b4, W5, b5, Wt2, Wt3, Wt4,
        Hh, Zh, acc, norm, dinv, Zs, acc1, E};

  // --- RK4 fixed-step integration: y'(t)=ode_f(y), 9 steps ---
  const float dt = (float)HOR / (float)(HOR - 1);
  k_store_pred<<<gblk(BN, TB), TB, 0, stream>>>(y, out, 0);
  for (int step = 1; step < HOR; ++step) {
    ode_f(c, y, k1, stream);
    k_axpy<<<gblk(BN, TB), TB, 0, stream>>>(y, k1, 0.5f * dt, ytmp);
    ode_f(c, ytmp, k2, stream);
    k_axpy<<<gblk(BN, TB), TB, 0, stream>>>(y, k2, 0.5f * dt, ytmp);
    ode_f(c, ytmp, k3, stream);
    k_axpy<<<gblk(BN, TB), TB, 0, stream>>>(y, k3, dt, ytmp);
    ode_f(c, ytmp, k4, stream);
    k_combine<<<gblk(BN, TB), TB, 0, stream>>>(y, k1, k2, k3, k4, dt / 6.f);
    k_store_pred<<<gblk(BN, TB), TB, 0, stream>>>(y, out, step);
  }
  (void)n_in; (void)out_size; (void)ws_size;
}
